// Attention_75935021793702
// MI455X (gfx1250) — matrix operand order repaired and checked by reading, not compiled, not run on hardware
//
#include <hip/hip_runtime.h>

// ---------------------------------------------------------------------------
// Types for CDNA5 WMMA (gfx1250, wave32)
// ---------------------------------------------------------------------------
typedef __attribute__((ext_vector_type(16))) __bf16 bf16x16;
typedef __attribute__((ext_vector_type(8)))  __bf16 bf16x8;
typedef __attribute__((ext_vector_type(8)))  float  floatx8;

static __device__ __forceinline__ __bf16 f2bf(float f) {
  unsigned u = __builtin_bit_cast(unsigned, f);
  u += 0x7FFFu + ((u >> 16) & 1u);          // round-to-nearest-even
  unsigned short h = (unsigned short)(u >> 16);
  return __builtin_bit_cast(__bf16, h);
}

static __device__ __forceinline__ unsigned pack2bf(float a, float b) {
  unsigned ua = __builtin_bit_cast(unsigned, a);
  unsigned ub = __builtin_bit_cast(unsigned, b);
  ua += 0x7FFFu + ((ua >> 16) & 1u);
  ub += 0x7FFFu + ((ub >> 16) & 1u);
  return (ua >> 16) | (ub & 0xFFFF0000u);
}

static __device__ __forceinline__ floatx8 zero8() {
  floatx8 z = {0.f, 0.f, 0.f, 0.f, 0.f, 0.f, 0.f, 0.f};
  return z;
}

// Async global->LDS 16-byte copy (CDNA5, tracked by ASYNCcnt).
static __device__ __forceinline__ void async_b128(unsigned lds_addr,
                                                  unsigned long long gaddr) {
  asm volatile("global_load_async_to_lds_b128 %0, %1, off"
               :: "v"(lds_addr), "v"(gaddr) : "memory");
}

// Problem constants (match the reference)
constexpr int Dmodel = 2048;
constexpr int Hheads = 16;
constexpr int DKc    = 128;
constexpr int Tc     = 2048;
constexpr int Bb     = 2;

// ---------------------------------------------------------------------------
// f32 -> bf16 streaming convert (paired for 32-bit stores).
// ---------------------------------------------------------------------------
__global__ __launch_bounds__(256) void convert_bf16(const float* __restrict__ in,
                                                    __bf16* __restrict__ out,
                                                    int npairs) {
  const int i = blockIdx.x * 256 + threadIdx.x;
  if (i >= npairs) return;
  const float2 v = *(const float2*)(in + 2 * (size_t)i);
  *(unsigned*)(out + 2 * (size_t)i) = pack2bf(v.x, v.y);
}

// ---------------------------------------------------------------------------
// W [K][N] f32  ->  Wt [N][K] bf16 (transpose + convert), 32x32 LDS tiles.
// ---------------------------------------------------------------------------
__global__ __launch_bounds__(256) void transpose_bf16(const float* __restrict__ W,
                                                      __bf16* __restrict__ Wt,
                                                      int K, int N) {
  __shared__ __bf16 tile[32][33];
  const int tx = threadIdx.x & 31, ty = threadIdx.x >> 5;   // 32 x 8
  const int gx = blockIdx.x * 32;  // N direction
  const int gy = blockIdx.y * 32;  // K direction
#pragma unroll
  for (int r = 0; r < 4; r++)
    tile[ty + 8 * r][tx] = f2bf(W[(size_t)(gy + ty + 8 * r) * N + gx + tx]);
  __syncthreads();
#pragma unroll
  for (int r = 0; r < 4; r++)
    Wt[(size_t)(gx + ty + 8 * r) * K + gy + tx] = tile[tx][ty + 8 * r];
}

// ---------------------------------------------------------------------------
// GEMM: C[M,N] f32 = A[M,K] bf16 * Bt[N,K] bf16 (B pre-transposed).
// Block 256 threads (8 waves), block tile 128x128, K-step 32, wave tile 32x64.
// LDS is double-buffered and stored in WMMA *fragment layout*, filled by
// global_load_async_to_lds_b128, so each fragment is one 32B LDS read.
// ---------------------------------------------------------------------------
__global__ __launch_bounds__(256) void gemm_bf16_async(
    const __bf16* __restrict__ A, const __bf16* __restrict__ Bt,
    float* __restrict__ C, int M, int N, int K) {
  // [buf][16-row tile][lane][element]
  __shared__ __align__(32) __bf16 AsSw[2][8][32][16];
  __shared__ __align__(32) __bf16 BsSw[2][8][32][16];

  const int tid  = threadIdx.x;
  const int wid  = tid >> 5;
  const int lane = tid & 31;
  const int lm   = lane & 15;
  const int hi   = lane >> 4;
  const int wave_m = wid >> 1;   // 0..3  (32 rows each)
  const int wave_n = wid & 1;    // 0..1  (64 cols each)
  const int m0 = blockIdx.y * 128;
  const int n0 = blockIdx.x * 128;

  floatx8 acc[2][4];
#pragma unroll
  for (int i = 0; i < 2; i++)
#pragma unroll
    for (int j = 0; j < 4; j++) acc[i][j] = zero8();

  // Stage one 128x32 A tile + 128x32 Bt tile into fragment-layout LDS.
  // 512 16B chunks per matrix; 2+2 async copies per thread.
  auto issue = [&](int bb, int k0) {
#pragma unroll
    for (int t = 0; t < 2; t++) {
      const int q = tid + t * 256;
      const int r = q >> 2, c = q & 3;          // row, k-chunk (8 elems)
      const unsigned long long ga =
          (unsigned long long)(size_t)(A + (size_t)(m0 + r) * K + k0 + c * 8);
      const unsigned la =
          (unsigned)(size_t)&AsSw[bb][r >> 4][(r & 15) + 16 * (c & 1)][8 * (c >> 1)];
      async_b128(la, ga);
    }
#pragma unroll
    for (int t = 0; t < 2; t++) {
      const int q = tid + t * 256;
      const int n = q >> 2, c = q & 3;
      const unsigned long long ga =
          (unsigned long long)(size_t)(Bt + (size_t)(n0 + n) * K + k0 + c * 8);
      const unsigned lb =
          (unsigned)(size_t)&BsSw[bb][n >> 4][(n & 15) + 16 * (c & 1)][8 * (c >> 1)];
      async_b128(lb, ga);
    }
  };

  const int nk = K / 32;
  issue(0, 0);
  for (int kt = 0; kt < nk; kt++) {
    const int bb = kt & 1;
    if (kt + 1 < nk) {
      issue(bb ^ 1, (kt + 1) * 32);                       // prefetch next tile
      asm volatile("s_wait_asynccnt 0x4" ::: "memory");   // current tile done
    } else {
      asm volatile("s_wait_asynccnt 0x0" ::: "memory");
    }
    __syncthreads();

    bf16x16 afr[2], bfr[4];
#pragma unroll
    for (int i = 0; i < 2; i++)
      afr[i] = *(const bf16x16*)&AsSw[bb][wave_m * 2 + i][lane][0];
#pragma unroll
    for (int j = 0; j < 4; j++)
      bfr[j] = *(const bf16x16*)&BsSw[bb][wave_n * 4 + j][lane][0];

#pragma unroll
    for (int i = 0; i < 2; i++)
#pragma unroll
      for (int j = 0; j < 4; j++)
        acc[i][j] = __builtin_amdgcn_wmma_f32_16x16x32_bf16(
            false, afr[i], false, bfr[j], (short)0, acc[i][j], false, false);
    __syncthreads();
  }

  // Store: C row = base + v + 8*hi, col = lane%16.
#pragma unroll
  for (int i = 0; i < 2; i++)
#pragma unroll
    for (int j = 0; j < 4; j++) {
      const int rbase = m0 + wave_m * 32 + i * 16 + hi * 8;
      const int col   = n0 + wave_n * 64 + j * 16 + lm;
#pragma unroll
      for (int v = 0; v < 8; v++)
        C[(size_t)(rbase + v) * N + col] = acc[i][j][v];
    }
}

// ---------------------------------------------------------------------------
// RoPE (rotate-half) in-place on f32 qkv scratch [B,T,3D].
// ---------------------------------------------------------------------------
__global__ __launch_bounds__(256) void rope_kernel(float* __restrict__ qkv,
                                                   const int* __restrict__ pos) {
  const int idx = blockIdx.x * blockDim.x + threadIdx.x;
  constexpr int total = Bb * Tc * Hheads * (DKc / 2);
  if (idx >= total) return;
  const int i = idx & 63;
  const int h = (idx >> 6) & (Hheads - 1);
  const int t = (idx >> 10) & (Tc - 1);
  const int b = idx >> 21;

  const float inv = __expf(-(float)(2 * i) * (9.210340371976184f / 128.f));
  const float ang = (float)pos[t] * inv;
  float sn, cs;
  __sincosf(ang, &sn, &cs);

  const size_t base = ((size_t)(b * Tc + t)) * (3 * Dmodel) + h * DKc;
  float q1 = qkv[base + i], q2 = qkv[base + i + 64];
  qkv[base + i]      = q1 * cs - q2 * sn;
  qkv[base + i + 64] = q1 * sn + q2 * cs;

  const size_t kb = base + Dmodel;
  float k1 = qkv[kb + i], k2 = qkv[kb + i + 64];
  qkv[kb + i]      = k1 * cs - k2 * sn;
  qkv[kb + i + 64] = k1 * sn + k2 * cs;
}

// ---------------------------------------------------------------------------
// Pack roped qkv (f32) into bf16 operand buffers:
//   qbf,kbf: [B*H][T][DK] row-major;  vbf: [B*H][DK][T] (transposed).
// ---------------------------------------------------------------------------
__global__ __launch_bounds__(256) void pack_qkv(const float* __restrict__ qkv,
                                                __bf16* __restrict__ qbf,
                                                __bf16* __restrict__ kbf,
                                                __bf16* __restrict__ vbf) {
  const int idx = blockIdx.x * 256 + threadIdx.x;   // over B*T*H*DK
  constexpr int total = Bb * Tc * Hheads * DKc;
  if (idx >= total) return;
  const int d = idx & (DKc - 1);
  const int h = (idx >> 7) & (Hheads - 1);
  const int t = (idx >> 11) & (Tc - 1);
  const int b = idx >> 22;

  const size_t src = (size_t)(b * Tc + t) * (3 * Dmodel) + h * DKc + d;
  const int bh = b * Hheads + h;
  qbf[((size_t)bh * Tc + t) * DKc + d] = f2bf(qkv[src]);
  kbf[((size_t)bh * Tc + t) * DKc + d] = f2bf(qkv[src + Dmodel]);
  vbf[((size_t)bh * DKc + d) * Tc + t] = f2bf(qkv[src + 2 * Dmodel]);
}

// ---------------------------------------------------------------------------
// Causal flash attention, bf16 operands, all fragments contiguous 32B loads.
// One wave = 16 query rows of one (b,h); 32-key tiles; online softmax in
// C-layout with 16-lane shfl reductions; P transposed C->A via per-wave LDS.
// Writes bf16 output [B][T][D] (A matrix of the final GEMM).
// ---------------------------------------------------------------------------
__global__ __launch_bounds__(256) void flash_attn_bf16(
    const __bf16* __restrict__ qbf, const __bf16* __restrict__ kbf,
    const __bf16* __restrict__ vbf, __bf16* __restrict__ ybf) {
  const int tid  = threadIdx.x;
  const int wid  = tid >> 5;
  const int lane = tid & 31;
  const int lm   = lane & 15;
  const int hi   = lane >> 4;
  const int h  = blockIdx.y;
  const int b  = blockIdx.z;
  const int bh = b * Hheads + h;
  const int q0 = blockIdx.x * 128 + wid * 16;

  __shared__ __bf16 Plds[8][16][32];   // per-wave private 16x32 tile

  const __bf16* qh = qbf + (size_t)bh * Tc * DKc;
  const __bf16* kh = kbf + (size_t)bh * Tc * DKc;
  const __bf16* vh = vbf + (size_t)bh * DKc * Tc;

  // Q as 4 A-fragments (16x32 covering DK=128): two 16B chunks each.
  bf16x16 qfr[4];
  const __bf16* qrow = qh + (size_t)(q0 + lm) * DKc;
#pragma unroll
  for (int kk = 0; kk < 4; kk++) {
    bf16x8 lo = *(const bf16x8*)(qrow + kk * 32 + hi * 16);
    bf16x8 hi8 = *(const bf16x8*)(qrow + kk * 32 + hi * 16 + 8);
    qfr[kk] = __builtin_shufflevector(lo, hi8, 0, 1, 2, 3, 4, 5, 6, 7, 8, 9,
                                      10, 11, 12, 13, 14, 15);
  }

  floatx8 o[8];
#pragma unroll
  for (int nt = 0; nt < 8; nt++) o[nt] = zero8();
  float mrow[8], lrow[8];
#pragma unroll
  for (int v = 0; v < 8; v++) { mrow[v] = -1e30f; lrow[v] = 0.f; }

  const float scale = 0.08838834764831845f;   // 1/sqrt(128)
  const int   nkt   = (q0 + 16 + 31) / 32;    // causal: skip above-diag tiles

  for (int kt = 0; kt < nkt; kt++) {
    const int kb0 = kt * 32;

    // S = Q K^T : K fragment = 32B contiguous (keys row-major in DK).
    floatx8 s[2] = {zero8(), zero8()};
#pragma unroll
    for (int kk = 0; kk < 4; kk++) {
#pragma unroll
      for (int sub = 0; sub < 2; sub++) {
        const bf16x16 kf = *(const bf16x16*)(
            kh + (size_t)(kb0 + sub * 16 + lm) * DKc + kk * 32 + hi * 16);
        s[sub] = __builtin_amdgcn_wmma_f32_16x16x32_bf16(
            false, qfr[kk], false, kf, (short)0, s[sub], false, false);
      }
    }

    // Scale + causal mask (row = q0 + v + 8*hi, key = kb0 + sub*16 + lane%16).
    const int qrow_off = q0 + hi * 8;
#pragma unroll
    for (int sub = 0; sub < 2; sub++) {
      const int key = kb0 + sub * 16 + lm;
#pragma unroll
      for (int v = 0; v < 8; v++) {
        const float sv = s[sub][v] * scale;
        s[sub][v] = (key <= qrow_off + v) ? sv : -1e30f;
      }
    }

    // Online softmax.
    float mnew[8], alpha[8];
#pragma unroll
    for (int v = 0; v < 8; v++) {
      float rm = fmaxf(s[0][v], s[1][v]);
#pragma unroll
      for (int off = 1; off < 16; off <<= 1)
        rm = fmaxf(rm, __shfl_xor(rm, off, 32));
      mnew[v]  = fmaxf(mrow[v], rm);
      alpha[v] = __expf(mrow[v] - mnew[v]);
      mrow[v]  = mnew[v];
    }
#pragma unroll
    for (int v = 0; v < 8; v++) {
      const float p0 = __expf(s[0][v] - mnew[v]);
      const float p1 = __expf(s[1][v] - mnew[v]);
      s[0][v] = p0; s[1][v] = p1;
      float rsum = p0 + p1;
#pragma unroll
      for (int off = 1; off < 16; off <<= 1)
        rsum += __shfl_xor(rsum, off, 32);
      lrow[v] = lrow[v] * alpha[v] + rsum;
    }
#pragma unroll
    for (int nt = 0; nt < 8; nt++)
#pragma unroll
      for (int v = 0; v < 8; v++) o[nt][v] *= alpha[v];

    // P: C-layout -> A-layout via per-wave LDS (in-order per wave).
#pragma unroll
    for (int sub = 0; sub < 2; sub++)
#pragma unroll
      for (int v = 0; v < 8; v++)
        Plds[wid][v + 8 * hi][sub * 16 + lm] = f2bf(s[sub][v]);
    __builtin_amdgcn_wave_barrier();
    bf16x16 pf;
    {
      bf16x8 lo = *(const bf16x8*)&Plds[wid][lm][hi * 16];
      bf16x8 hi8 = *(const bf16x8*)&Plds[wid][lm][hi * 16 + 8];
      pf = __builtin_shufflevector(lo, hi8, 0, 1, 2, 3, 4, 5, 6, 7, 8, 9, 10,
                                   11, 12, 13, 14, 15);
    }
    __builtin_amdgcn_wave_barrier();

    // O += P V : V fragment = 32B contiguous (V stored [DK][T]).
#pragma unroll
    for (int nt = 0; nt < 8; nt++) {
      const bf16x16 vf = *(const bf16x16*)(
          vh + (size_t)(nt * 16 + lm) * Tc + kb0 + hi * 16);
      o[nt] = __builtin_amdgcn_wmma_f32_16x16x32_bf16(
          false, pf, false, vf, (short)0, o[nt], false, false);
    }
  }

  // Epilogue: O / l -> bf16 y[b, q0+row, h*DK + col].
  __bf16* yrow = ybf + ((size_t)b * Tc + q0) * Dmodel + h * DKc;
#pragma unroll
  for (int nt = 0; nt < 8; nt++)
#pragma unroll
    for (int v = 0; v < 8; v++)
      yrow[(size_t)(v + 8 * hi) * Dmodel + nt * 16 + lm] = f2bf(o[nt][v] / lrow[v]);
}

// ---------------------------------------------------------------------------
// Launcher. Inputs: x, mask(unused: causal analytic), pos, Wqkv, Wo.
// ---------------------------------------------------------------------------
extern "C" void kernel_launch(void* const* d_in, const int* in_sizes, int n_in,
                              void* d_out, int out_size, void* d_ws, size_t ws_size,
                              hipStream_t stream) {
  (void)in_sizes; (void)n_in; (void)out_size; (void)ws_size;
  const float* x    = (const float*)d_in[0];
  const int*   pos  = (const int*)d_in[2];
  const float* Wqkv = (const float*)d_in[3];
  const float* Wo   = (const float*)d_in[4];
  float*       out  = (float*)d_out;

  const int M = Bb * Tc;   // 4096

  // Workspace layout (bytes, all sub-buffers 256B-aligned by construction).
  char* ws = (char*)d_ws;
  float* qkv = (float*)ws;            ws += (size_t)M * 3 * Dmodel * 4;   // 100.7MB
  __bf16* xbf   = (__bf16*)ws;        ws += (size_t)M * Dmodel * 2;       // 16.8MB
  __bf16* wqkvt = (__bf16*)ws;        ws += (size_t)3 * Dmodel * Dmodel * 2; // 25.2MB
  __bf16* wot   = (__bf16*)ws;        ws += (size_t)Dmodel * Dmodel * 2;  // 8.4MB
  __bf16* qb    = (__bf16*)ws;        ws += (size_t)M * Dmodel * 2;
  __bf16* kb    = (__bf16*)ws;        ws += (size_t)M * Dmodel * 2;
  __bf16* vb    = (__bf16*)ws;        ws += (size_t)M * Dmodel * 2;
  __bf16* yb    = (__bf16*)ws;        ws += (size_t)M * Dmodel * 2;

  // 1) Convert x to bf16; transpose+convert weights to [N][K] bf16.
  {
    const int npairs = M * Dmodel / 2;
    convert_bf16<<<(npairs + 255) / 256, 256, 0, stream>>>(x, xbf, npairs);
    dim3 g1((3 * Dmodel) / 32, Dmodel / 32);
    transpose_bf16<<<g1, 256, 0, stream>>>(Wqkv, wqkvt, Dmodel, 3 * Dmodel);
    dim3 g2(Dmodel / 32, Dmodel / 32);
    transpose_bf16<<<g2, 256, 0, stream>>>(Wo, wot, Dmodel, Dmodel);
  }
  // 2) qkv = x @ Wqkv (f32 accumulate/out).
  {
    dim3 grid((3 * Dmodel) / 128, M / 128);
    gemm_bf16_async<<<grid, 256, 0, stream>>>(xbf, wqkvt, qkv, M, 3 * Dmodel, Dmodel);
  }
  // 3) RoPE in place on q,k.
  {
    const int total = Bb * Tc * Hheads * (DKc / 2);
    rope_kernel<<<(total + 255) / 256, 256, 0, stream>>>(qkv, pos);
  }
  // 4) Pack q/k/v into bf16 fragment-friendly buffers (V transposed).
  {
    const int total = Bb * Tc * Hheads * DKc;
    pack_qkv<<<(total + 255) / 256, 256, 0, stream>>>(qkv, qb, kb, vb);
  }
  // 5) Causal flash attention -> bf16 y.
  {
    dim3 grid(Tc / 128, Hheads, Bb);
    flash_attn_bf16<<<grid, 256, 0, stream>>>(qb, kb, vb, yb);
  }
  // 6) out = y @ Wo.
  {
    dim3 grid(Dmodel / 128, M / 128);
    gemm_bf16_async<<<grid, 256, 0, stream>>>(yb, wot, out, M, Dmodel, Dmodel);
  }
}
